// MultiHeadAttention_77910706749565
// MI455X (gfx1250) — matrix operand order repaired and checked by reading, not compiled, not run on hardware
//
#include <hip/hip_runtime.h>
#include <hip/hip_bf16.h>

typedef __attribute__((ext_vector_type(16))) _Float16 v16h;
typedef __attribute__((ext_vector_type(8)))  _Float16 v8h;
typedef __attribute__((ext_vector_type(8)))  float    v8f;

#define B_  2
#define L_  4096
#define D_  512
#define H_  8
#define DH_ 64

__device__ __forceinline__ v8f wmma16(v16h a, v16h b, v8f c) {
  // D = A(16x32 f16) * B(32x16 f16) + C(16x16 f32)
  return __builtin_amdgcn_wmma_f32_16x16x32_f16(false, a, false, b, (short)0, c,
                                                false, false);
}

// Load 8 consecutive f32 and convert to f16.
__device__ __forceinline__ void cvt8(const float* __restrict__ src, _Float16* dst) {
  float4 p0 = ((const float4*)src)[0];
  float4 p1 = ((const float4*)src)[1];
  dst[0] = (_Float16)p0.x; dst[1] = (_Float16)p0.y;
  dst[2] = (_Float16)p0.z; dst[3] = (_Float16)p0.w;
  dst[4] = (_Float16)p1.x; dst[5] = (_Float16)p1.y;
  dst[6] = (_Float16)p1.z; dst[7] = (_Float16)p1.w;
}

// A-matrix fragment (16x32 f16): per lane, 16 consecutive k values at base.
__device__ __forceinline__ v16h afrag_f32(const float* __restrict__ base) {
  _Float16 t[16];
  cvt8(base, t);
  cvt8(base + 8, t + 8);
  v16h a;
#pragma unroll
  for (int i = 0; i < 16; ++i) a[i] = t[i];
  return a;
}

// B-matrix fragment (32x16 f16) from f32 source: 16 consecutive values.
__device__ __forceinline__ v16h bfrag_f32(const float* __restrict__ base) {
  _Float16 t[16];
  cvt8(base, t);
  cvt8(base + 8, t + 8);
  v16h b;
#pragma unroll
  for (int i = 0; i < 16; ++i) b[i] = t[i];
  return b;
}

__device__ __forceinline__ v16h afrag_f16(const _Float16* __restrict__ base) {
  v8h c0 = *(const v8h*)(base);
  v8h c1 = *(const v8h*)(base + 8);
  v16h a;
#pragma unroll
  for (int i = 0; i < 8; ++i) { a[i] = c0[i]; a[8 + i] = c1[i]; }
  return a;
}

__device__ __forceinline__ v16h bfrag_f16(const _Float16* __restrict__ base) {
  return *(const v16h*)base;  // 16 consecutive f16 (32B)
}

// C = X @ W^T + bias. X: [8192, 512] f32, W: [512, 512] f32.
// mode 0/1: write f16 head-split [B,H,L,Dh] * scale.  mode 2: write f32 [M,512].
// Each wave computes a 32x32 output tile.
__global__ void proj_kernel(const float* __restrict__ X, const float* __restrict__ W,
                            const float* __restrict__ bias,
                            _Float16* __restrict__ outH, float* __restrict__ outF,
                            int mode, float scale) {
  const int lane = threadIdx.x & 31;
  const int wave = threadIdx.x >> 5;
  const int tile = blockIdx.x * (blockDim.x >> 5) + wave;
  const int m0 = (tile >> 4) * 32;   // 256 M-tiles of 32
  const int n0 = (tile & 15) * 32;   // 16  N-tiles of 32
  const int lm = lane & 15, lhi = lane >> 4;

  v8f acc[2][2] = {};
  for (int kc = 0; kc < D_; kc += 32) {
    if (kc + 32 < D_) {  // prefetch next k-step A rows (global_prefetch_b8)
      __builtin_prefetch(X + (size_t)(m0 + lm) * D_ + kc + 32 + 8 * lhi, 0, 3);
      __builtin_prefetch(X + (size_t)(m0 + 16 + lm) * D_ + kc + 32 + 8 * lhi, 0, 3);
    }
    v16h a[2], b[2];
#pragma unroll
    for (int mi = 0; mi < 2; ++mi)
      a[mi] = afrag_f32(X + (size_t)(m0 + 16 * mi + lm) * D_ + kc + 16 * lhi);
#pragma unroll
    for (int ni = 0; ni < 2; ++ni)
      b[ni] = bfrag_f32(W + (size_t)(n0 + 16 * ni + lm) * D_ + kc + 16 * lhi);
#pragma unroll
    for (int mi = 0; mi < 2; ++mi)
#pragma unroll
      for (int ni = 0; ni < 2; ++ni)
        acc[mi][ni] = wmma16(a[mi], b[ni], acc[mi][ni]);
  }

#pragma unroll
  for (int mi = 0; mi < 2; ++mi) {
#pragma unroll
    for (int ni = 0; ni < 2; ++ni) {
      const int n = n0 + 16 * ni + lm;
      const float bv = bias[n];
#pragma unroll
      for (int r = 0; r < 8; ++r) {
        const int m = m0 + 16 * mi + 8 * lhi + r;
        const float v = acc[mi][ni][r] + bv;
        if (mode == 2) {
          outF[(size_t)m * D_ + n] = v;
        } else {
          const int bb = m >> 12, lrow = m & (L_ - 1);
          const int h = n >> 6, dh = n & 63;
          outH[(((size_t)(bb * H_ + h) * L_) + lrow) * DH_ + dh] =
              (_Float16)(v * scale);
        }
      }
    }
  }
}

// Raw scores: S[b,h,i,j] = dot(Qh[i,:], Kh[j,:])   (scale folded into Qh)
// Per (b,h): GEMM 4096x4096x64. Each wave: 32x64 tile (8 WMMA per k-step).
__global__ void scores_kernel(const _Float16* __restrict__ Qh,
                              const _Float16* __restrict__ Kh,
                              float* __restrict__ attn) {
  const int lane = threadIdx.x & 31;
  const int wave = threadIdx.x >> 5;
  const int tile = blockIdx.x * (blockDim.x >> 5) + wave;
  const int bh  = tile >> 13;        // 128 * 64 tiles per (b,h)
  const int rem = tile & 8191;
  const int m0 = (rem >> 6) * 32;    // 128 M-tiles of 32
  const int n0 = (rem & 63) * 64;    // 64  N-tiles of 64
  const int lm = lane & 15, lhi = lane >> 4;
  const _Float16* Q = Qh + (size_t)bh * L_ * DH_;
  const _Float16* K = Kh + (size_t)bh * L_ * DH_;

  v8f acc[2][4] = {};
#pragma unroll
  for (int kc = 0; kc < DH_; kc += 32) {
    v16h a[2], b[4];
#pragma unroll
    for (int mi = 0; mi < 2; ++mi)
      a[mi] = afrag_f16(Q + (size_t)(m0 + 16 * mi + lm) * DH_ + kc + 16 * lhi);
#pragma unroll
    for (int ni = 0; ni < 4; ++ni)
      b[ni] = bfrag_f16(K + (size_t)(n0 + 16 * ni + lm) * DH_ + kc + 16 * lhi);
#pragma unroll
    for (int mi = 0; mi < 2; ++mi)
#pragma unroll
      for (int ni = 0; ni < 4; ++ni)
        acc[mi][ni] = wmma16(a[mi], b[ni], acc[mi][ni]);
  }

  float* S = attn + (size_t)bh * L_ * L_;
#pragma unroll
  for (int mi = 0; mi < 2; ++mi)
#pragma unroll
    for (int ni = 0; ni < 4; ++ni)
#pragma unroll
      for (int r = 0; r < 8; ++r)
        S[(size_t)(m0 + 16 * mi + 8 * lhi + r) * L_ + (n0 + 16 * ni + lm)] =
            acc[mi][ni][r];
}

// In-place row softmax: one 256-thread block per row of 4096.
__global__ void softmax_kernel(float* __restrict__ attn) {
  const int tid = threadIdx.x;
  float* row = attn + (size_t)blockIdx.x * L_;
  float v[16];
#pragma unroll
  for (int i = 0; i < 16; ++i) v[i] = row[tid + 256 * i];

  __shared__ float red[256];
  float mx = v[0];
#pragma unroll
  for (int i = 1; i < 16; ++i) mx = fmaxf(mx, v[i]);
  red[tid] = mx;
  __syncthreads();
  for (int s = 128; s > 0; s >>= 1) {
    if (tid < s) red[tid] = fmaxf(red[tid], red[tid + s]);
    __syncthreads();
  }
  mx = red[0];
  __syncthreads();

  float sum = 0.f;
#pragma unroll
  for (int i = 0; i < 16; ++i) { v[i] = __expf(v[i] - mx); sum += v[i]; }
  red[tid] = sum;
  __syncthreads();
  for (int s = 128; s > 0; s >>= 1) {
    if (tid < s) red[tid] += red[tid + s];
    __syncthreads();
  }
  const float inv = 1.0f / red[0];
#pragma unroll
  for (int i = 0; i < 16; ++i) row[tid + 256 * i] = v[i] * inv;
}

// context[b,l,h*64+d] = sum_j attn[b,h,l,j] * Vh[b,h,j,d]
// Block (8 waves) covers 128 rows x 64 cols; V tile staged transposed in LDS.
__global__ void context_kernel(const float* __restrict__ attn,
                               const _Float16* __restrict__ Vh,
                               float* __restrict__ ctx) {
  const int tid = threadIdx.x;
  const int lane = tid & 31, wave = tid >> 5;
  const int bh = blockIdx.x >> 5;
  const int m0 = (blockIdx.x & 31) * 128 + wave * 16;
  const int lm = lane & 15, lhi = lane >> 4;
  const int b = bh >> 3, h = bh & 7;
  const float* A = attn + (size_t)bh * L_ * L_;
  const _Float16* V = Vh + (size_t)bh * L_ * DH_;

  __shared__ _Float16 vt[DH_ * 32];  // transposed: vt[d*32 + j_local]
  const int jl = tid >> 3;           // 0..31
  const int d0 = (tid & 7) * 8;      // 0..56

  v8f acc[4] = {};
  for (int kc = 0; kc < L_; kc += 32) {
    if (kc + 32 < L_) {  // prefetch next attn A rows + V tile
      __builtin_prefetch(A + (size_t)(m0 + lm) * L_ + kc + 32 + 8 * lhi, 0, 3);
      __builtin_prefetch(V + (size_t)(kc + 32 + jl) * DH_ + d0, 0, 3);
    }
    v8h vv = *(const v8h*)(V + (size_t)(kc + jl) * DH_ + d0);
#pragma unroll
    for (int i = 0; i < 8; ++i) vt[(d0 + i) * 32 + jl] = vv[i];
    __syncthreads();

    v16h a = afrag_f32(A + (size_t)(m0 + lm) * L_ + kc + 16 * lhi);
#pragma unroll
    for (int t = 0; t < 4; ++t) {
      v16h bf = *(const v16h*)(&vt[(16 * t + lm) * 32 + 16 * lhi]);
      acc[t] = wmma16(a, bf, acc[t]);
    }
    __syncthreads();
  }

#pragma unroll
  for (int t = 0; t < 4; ++t)
#pragma unroll
    for (int r = 0; r < 8; ++r) {
      const int m = m0 + 8 * lhi + r;
      const int d = 16 * t + lm;
      ctx[((size_t)b * L_ + m) * D_ + h * DH_ + d] = acc[t][r];
    }
}

extern "C" void kernel_launch(void* const* d_in, const int* in_sizes, int n_in,
                              void* d_out, int out_size, void* d_ws, size_t ws_size,
                              hipStream_t stream) {
  const float* q  = (const float*)d_in[0];
  const float* k  = (const float*)d_in[1];
  const float* v  = (const float*)d_in[2];
  const float* Wq = (const float*)d_in[3];
  const float* bq = (const float*)d_in[4];
  const float* Wk = (const float*)d_in[5];
  const float* bk = (const float*)d_in[6];
  const float* Wv = (const float*)d_in[7];
  const float* bv = (const float*)d_in[8];
  const float* Wo = (const float*)d_in[9];
  const float* bo = (const float*)d_in[10];

  float* out  = (float*)d_out;                        // [2,4096,512]
  float* attn = (float*)d_out + (size_t)B_ * L_ * D_; // [2,8,4096,4096]

  char* ws = (char*)d_ws;                             // 40 MB used
  _Float16* Qh = (_Float16*)(ws);                     //  8 MB f16 [B,H,L,64]
  _Float16* Kh = (_Float16*)(ws + (8u << 20));        //  8 MB
  _Float16* Vh = (_Float16*)(ws + (16u << 20));       //  8 MB
  float*    ctx = (float*)(ws + (24u << 20));         // 16 MB f32 [B,L,512]

  // Q/K/V projections (scale 1/sqrt(64)=0.125 folded into Q)
  proj_kernel<<<1024, 128, 0, stream>>>(q, Wq, bq, Qh, nullptr, 1, 0.125f);
  proj_kernel<<<1024, 128, 0, stream>>>(k, Wk, bk, Kh, nullptr, 0, 1.0f);
  proj_kernel<<<1024, 128, 0, stream>>>(v, Wv, bv, Vh, nullptr, 0, 1.0f);
  // Raw scores into attn region of d_out  (131072 wave-tiles / 4 waves per block)
  scores_kernel<<<32768, 128, 0, stream>>>(Qh, Kh, attn);
  // Row softmax in place
  softmax_kernel<<<65536, 256, 0, stream>>>(attn);
  // context = attn @ V, regathered to [B,L,512]
  context_kernel<<<512, 256, 0, stream>>>(attn, Vh, ctx);
  // Output projection
  proj_kernel<<<1024, 128, 0, stream>>>(ctx, Wo, bo, nullptr, out, 2, 1.0f);
}
